// RelativePositionalMultiHeadAttn_37993280700957
// MI455X (gfx1250) — compile-verified, matrix operand order checked by reading, not run on hardware
//
#include <hip/hip_runtime.h>
#include <hip/hip_bf16.h>

typedef __attribute__((ext_vector_type(16))) _Float16 v16h;
typedef __attribute__((ext_vector_type(8)))  float    v8f;
typedef _Float16 h8_t __attribute__((ext_vector_type(8)));
typedef _Float16 h4_t __attribute__((ext_vector_type(4)));

#define SEQ 49
#define DPAD 64      // 49 padded to 64 rows (4 M-tiles)
#define FEAT 128
#define NH 4
#define HD 32
#define NEG_BIG (-1e30f)

// ---------------------------------------------------------------------------
// Fragment helpers (wave32, 16x16x32 f16 WMMA)
// A operand (16x32, MxK), per ISA 7.12.2: lane L row m=L&15; K-halves:
//   kh = (L>=16)?8:0 ; VGPR v<4 -> K = kh+2v ; v>=4 -> K = 16+kh+2(v-4)
// => per lane: 8 contiguous halves at K=kh and 8 at K=16+kh -> two b128 loads.
// B operand (32x16, KxN) mirrors this with lane -> column n, so loading from
// an N-major (K-contiguous) buffer uses the identical access pattern.
// ---------------------------------------------------------------------------
__device__ __forceinline__ v16h load_frag(const _Float16* base, int ld) {
  const int lane = threadIdx.x & 31;
  const int r  = lane & 15;
  const int kh = (lane >> 4) << 3;           // 0 or 8
  const _Float16* p = base + r * ld + kh;
  h8_t lo = *(const h8_t*)(p);
  h8_t hi = *(const h8_t*)(p + 16);
  v16h out;
#pragma unroll
  for (int i = 0; i < 8; ++i) { out[i] = lo[i]; out[i + 8] = hi[i]; }
  return out;
}

// C/D layout: VGPR r -> m = ((lane>=16)?8:0)+r, n = lane&15
__device__ __forceinline__ void store_c_f16(_Float16* dst, int ld, v8f c,
                                            float bias, float scale) {
  const int lane = threadIdx.x & 31;
  const int n  = lane & 15;
  const int mh = (lane >> 4) << 3;
#pragma unroll
  for (int r = 0; r < 8; ++r)
    dst[(mh + r) * ld + n] = (_Float16)((c[r] + bias) * scale);
}

__device__ __forceinline__ void store_c_f16_colmajor(_Float16* dst, int ld, v8f c,
                                                     float bias) {
  const int lane = threadIdx.x & 31;
  const int n  = lane & 15;
  const int mh = (lane >> 4) << 3;
  h8_t h;
#pragma unroll
  for (int r = 0; r < 8; ++r) h[r] = (_Float16)(c[r] + bias);
  *(h8_t*)(dst + n * ld + mh) = h;           // one b128 store per lane
}

__device__ __forceinline__ void store_c_f32(float* dst, int ld, v8f c) {
  const int lane = threadIdx.x & 31;
  const int n  = lane & 15;
  const int mh = (lane >> 4) << 3;
#pragma unroll
  for (int r = 0; r < 8; ++r) dst[(mh + r) * ld + n] = c[r];
}

#define WMMA_F16(A, B, C) \
  __builtin_amdgcn_wmma_f32_16x16x32_f16(false, (A), false, (B), (short)0, (C), false, false)

// ---------------------------------------------------------------------------
// Prep: (a) transpose+convert weights to f16, (b) build full relative-position
// bias matrix biasG[h][64][64] with -1e30 in padded slots (masks softmax).
// All outputs are small and L2-resident (192 MB L2), shared by all 16384 WGs.
// ---------------------------------------------------------------------------
__global__ void prep_weights(const float* __restrict__ Wqkv,
                             const float* __restrict__ Wmerge,
                             const float* __restrict__ table,
                             _Float16* __restrict__ WqkvT,
                             _Float16* __restrict__ WmergeT,
                             float* __restrict__ biasG) {
  int tid = blockIdx.x * blockDim.x + threadIdx.x;
  if (tid < 128 * 384) {
    int k = tid / 384, n = tid % 384;
    WqkvT[n * 128 + k] = (_Float16)Wqkv[tid];
  }
  if (tid < 128 * 128) {
    int k = tid / 128, n = tid % 128;
    WmergeT[n * 128 + k] = (_Float16)Wmerge[tid];
  }
  if (tid < NH * DPAD * DPAD) {
    int h = tid >> 12;
    int ij = tid & 4095;
    int i = ij >> 6, j = ij & 63;
    float v = NEG_BIG;
    if (i < SEQ && j < SEQ) {
      int ri = i / 7, ci = i - ri * 7;
      int rj = j / 7, cj = j - rj * 7;
      int idx = (ri - rj + 6) * 13 + (ci - cj + 6);
      v = table[idx * NH + h];
    }
    biasG[tid] = v;
  }
}

// ---------------------------------------------------------------------------
// Fused window attention: one workgroup per (b,g) window, 8 waves.
// ---------------------------------------------------------------------------
__global__ __launch_bounds__(256) void window_attn_kernel(
    const float* __restrict__ x, const float* __restrict__ bqkv,
    const float* __restrict__ biasG, const float* __restrict__ bmerge,
    const _Float16* __restrict__ WqkvT, const _Float16* __restrict__ WmergeT,
    float* __restrict__ out) {

  __shared__ __align__(16) _Float16 xA[DPAD * FEAT];     // also reused as `merged`
  __shared__ __align__(16) _Float16 q_lds[DPAD * FEAT];
  __shared__ __align__(16) _Float16 k_lds[DPAD * FEAT];
  __shared__ __align__(16) _Float16 vT[FEAT * DPAD];     // v transposed: [feat][token]
  __shared__ __align__(16) float    s_lds[DPAD * DPAD];  // attn logits (per head)
  __shared__ __align__(16) _Float16 p_lds[DPAD * DPAD];  // softmax probs (per head)

  const int bg   = blockIdx.x;
  const int tid  = threadIdx.x;
  const int wave = tid >> 5;
  const float scale = 0.17677669529663687f;   // 1/sqrt(32)

  // Warm L2/L0 for the weight images used by every workgroup.
  __builtin_prefetch(WqkvT + tid * 64, 0, 1);
  __builtin_prefetch(WmergeT + tid * 64, 0, 1);

  // ---- load x tile (49x128 f32 -> 64x128 f16, zero padded), float4 path ----
  const float4* xp4 = (const float4*)(x + (size_t)bg * SEQ * FEAT);
  for (int i = tid; i < DPAD * FEAT / 4; i += 256) {   // 8 iterations
    int row = i >> 5;                                  // 32 float4 per row
    h4_t hv;
    if (row < SEQ) {
      float4 v = xp4[i];
      hv[0] = (_Float16)v.x; hv[1] = (_Float16)v.y;
      hv[2] = (_Float16)v.z; hv[3] = (_Float16)v.w;
    } else {
      hv[0] = hv[1] = hv[2] = hv[3] = (_Float16)0.f;
    }
    *(h4_t*)(xA + i * 4) = hv;
  }
  __syncthreads();

  // ---- QKV projection: [64x128] @ [128x384] -> q,k (row-major) and v (transposed)
  for (int t = wave; t < 96; t += 8) {          // 4 M-tiles x 24 N-tiles
    const int mt = t & 3, nt = t >> 2;
    v8f acc = {};
#pragma unroll
    for (int kk = 0; kk < 4; ++kk) {
      v16h a = load_frag(xA + mt * 16 * FEAT + kk * 32, FEAT);
      v16h b = load_frag(WqkvT + nt * 16 * FEAT + kk * 32, FEAT);
      acc = WMMA_F16(a, b, acc);
    }
    const int ng = nt * 16 + (tid & 15);        // column in 0..383
    const float bias = bqkv[ng];
    if (nt < 8) {
      store_c_f16(q_lds + mt * 16 * FEAT + nt * 16, FEAT, acc, bias, scale);
    } else if (nt < 16) {
      store_c_f16(k_lds + mt * 16 * FEAT + (nt - 8) * 16, FEAT, acc, bias, 1.f);
    } else {
      store_c_f16_colmajor(vT + (nt - 16) * 16 * DPAD + mt * 16, DPAD, acc, bias);
    }
  }
  __syncthreads();

  _Float16* merged = xA;                        // xA no longer needed -> reuse

  for (int h = 0; h < NH; ++h) {
    // ---- logits = (q*scale) @ k^T : 4x4 tiles, K=32 (one WMMA each) ----
    for (int t = wave; t < 16; t += 8) {
      const int mt = t & 3, nt = t >> 2;
      v16h a = load_frag(q_lds + mt * 16 * FEAT + h * HD, FEAT);
      v16h b = load_frag(k_lds + nt * 16 * FEAT + h * HD, FEAT);  // row-major k == k^T B-operand
      v8f acc = {};
      acc = WMMA_F16(a, b, acc);
      store_c_f32(s_lds + mt * 16 * DPAD + nt * 16, DPAD, acc);
    }
    __syncthreads();

    // ---- softmax + bias: 4 lanes per row, 16 cols each, shuffle reductions.
    // biasG already holds -1e30 in every padded (i>=49 or j>=49) slot, so the
    // masking is pure arithmetic: exp(-1e30 - mx) == 0.
    {
      const int i  = tid >> 2;                  // row 0..63
      const int q4 = tid & 3;                   // column quarter
      float* row = s_lds + i * DPAD + q4 * 16;
      const float* bp = biasG + h * DPAD * DPAD + i * DPAD + q4 * 16;
      float mx = NEG_BIG;
#pragma unroll
      for (int j = 0; j < 16; ++j) {
        float v = row[j] + bp[j];
        row[j] = v;
        mx = fmaxf(mx, v);
      }
      mx = fmaxf(mx, __shfl_xor(mx, 1, 32));
      mx = fmaxf(mx, __shfl_xor(mx, 2, 32));
      float sum = 0.f;
#pragma unroll
      for (int j = 0; j < 16; ++j) {
        float e = __expf(row[j] - mx);
        row[j] = e;
        sum += e;
      }
      sum += __shfl_xor(sum, 1, 32);
      sum += __shfl_xor(sum, 2, 32);
      const float rinv = 1.f / sum;
      h8_t pa, pb;
#pragma unroll
      for (int j = 0; j < 8; ++j) {
        pa[j] = (_Float16)(row[j] * rinv);
        pb[j] = (_Float16)(row[j + 8] * rinv);
      }
      _Float16* pd = p_lds + i * DPAD + q4 * 16;
      *(h8_t*)(pd)     = pa;                    // two b128 stores
      *(h8_t*)(pd + 8) = pb;
    }
    __syncthreads();

    // ---- out_h = P[64x64] @ v_h[64x32] : 4x2 tiles, one per wave, K=64 ----
    {
      const int mt = wave & 3, nt = wave >> 2;
      v8f acc = {};
#pragma unroll
      for (int kk = 0; kk < 2; ++kk) {
        v16h a = load_frag(p_lds + mt * 16 * DPAD + kk * 32, DPAD);
        v16h b = load_frag(vT + (h * HD + nt * 16) * DPAD + kk * 32, DPAD);
        acc = WMMA_F16(a, b, acc);
      }
      store_c_f16(merged + mt * 16 * FEAT + h * HD + nt * 16, FEAT, acc, 0.f, 1.f);
    }
    __syncthreads();
  }

  // ---- final merge: [64x128] @ [128x128] + bmerge -> global f32 ----
  const size_t obase = (size_t)bg * SEQ * FEAT;
  const int mh = ((tid >> 4) & 1) << 3;
  for (int t = wave; t < 32; t += 8) {          // 4 M-tiles x 8 N-tiles
    const int mt = t & 3, nt = t >> 2;
    v8f acc = {};
#pragma unroll
    for (int kk = 0; kk < 4; ++kk) {
      v16h a = load_frag(merged + mt * 16 * FEAT + kk * 32, FEAT);
      v16h b = load_frag(WmergeT + nt * 16 * FEAT + kk * 32, FEAT);
      acc = WMMA_F16(a, b, acc);
    }
    const int n = nt * 16 + (tid & 15);
    const float bias = bmerge[n];
#pragma unroll
    for (int r = 0; r < 8; ++r) {
      const int m = mt * 16 + mh + r;
      if (m < SEQ) out[obase + (size_t)m * FEAT + n] = acc[r] + bias;
    }
  }
}

extern "C" void kernel_launch(void* const* d_in, const int* in_sizes, int n_in,
                              void* d_out, int out_size, void* d_ws, size_t ws_size,
                              hipStream_t stream) {
  (void)in_sizes; (void)n_in; (void)out_size; (void)ws_size;
  const float* x      = (const float*)d_in[0];
  const float* Wqkv   = (const float*)d_in[1];
  const float* bqkv   = (const float*)d_in[2];
  const float* Wmerge = (const float*)d_in[3];
  const float* bmerge = (const float*)d_in[4];
  const float* table  = (const float*)d_in[5];

  _Float16* WqkvT   = (_Float16*)d_ws;                 // 384*128 halves (96 KB)
  _Float16* WmergeT = WqkvT + 384 * 128;               // 128*128 halves (32 KB)
  float*    biasG   = (float*)(WmergeT + 128 * 128);   // 4*64*64 floats (64 KB)

  prep_weights<<<(128 * 384 + 255) / 256, 256, 0, stream>>>(
      Wqkv, Wmerge, table, WqkvT, WmergeT, biasG);

  const int nblocks = 16 * 1024;                       // B * G windows
  window_attn_kernel<<<nblocks, 256, 0, stream>>>(x, bqkv, biasG, bmerge,
                                                  WqkvT, WmergeT, (float*)d_out);
}
